// GCN_Estimator_39462159515954
// MI455X (gfx1250) — matrix operand order repaired and checked by reading, not compiled, not run on hardware
//
#include <hip/hip_runtime.h>
#include <hip/hip_bf16.h>

#define N_USER   150000
#define M_ITEM   30000
#define N_NODES  (N_USER + M_ITEM)
#define DD       64
#define B_ROWS   1024
#define AUX_STRIDE 10246
#define X_COLS   640

#define K_GENRE 25
#define K_DIR   2186
#define K_ACT   8030
#define KP_GENRE 32
#define KP_DIR   2208
#define KP_ACT   8032

typedef __attribute__((ext_vector_type(16))) _Float16 v16h;
typedef __attribute__((ext_vector_type(8)))  _Float16 v8h;
typedef __attribute__((ext_vector_type(8)))  float    v8f;

// ---------------------------------------------------------------------
// WMMA fragment layouts as used here (contiguous k order, 16-bit data), wave32:
//  A 16x32: lane L -> row M=L%16, g=L/16. elem e    : K = k0 + 16g + e
//  B 32x16: lane L -> col N=L%16, g=L/16. elem e    : K = k0 + 16g + e
//  C/D    : elem r -> M = r + 8g, N = L%16
// => with zero-padded f16 operands, every fragment is aligned b128 loads.
// ---------------------------------------------------------------------

__device__ __forceinline__ v16h cat8(v8h lo, v8h hi) {
  return __builtin_shufflevector(lo, hi, 0,1,2,3,4,5,6,7,8,9,10,11,12,13,14,15);
}

// ---------------------------------------------------------------------
// Pack 0/1 aux sections -> zero-padded f16 matrices + row sums
// ---------------------------------------------------------------------
__global__ __launch_bounds__(256)
void pack_aux_kernel(const int* __restrict__ aux,
                     _Float16* __restrict__ Ag, _Float16* __restrict__ Ad,
                     _Float16* __restrict__ Aa,
                     float* __restrict__ rsg, float* __restrict__ rsd,
                     float* __restrict__ rsa) {
  int b = blockIdx.x;
  int t = threadIdx.x;
  const int* arow = aux + (long)b * AUX_STRIDE;
  __shared__ int s_sum[3];
  if (t < 3) s_sum[t] = 0;
  __syncthreads();
  int local = 0;
  for (int k = t; k < KP_GENRE; k += 256) {
    int v = (k < K_GENRE) ? arow[1 + k] : 0;
    Ag[(long)b * KP_GENRE + k] = (_Float16)v;  local += v;
  }
  atomicAdd(&s_sum[0], local);  local = 0;
  for (int k = t; k < KP_DIR; k += 256) {
    int v = (k < K_DIR) ? arow[26 + k] : 0;
    Ad[(long)b * KP_DIR + k] = (_Float16)v;  local += v;
  }
  atomicAdd(&s_sum[1], local);  local = 0;
  for (int k = t; k < KP_ACT; k += 256) {
    int v = (k < K_ACT) ? arow[2212 + k] : 0;
    Aa[(long)b * KP_ACT + k] = (_Float16)v;  local += v;
  }
  atomicAdd(&s_sum[2], local);
  __syncthreads();
  if (t == 0) {
    rsg[b] = (float)s_sum[0];
    rsd[b] = (float)s_sum[1];
    rsa[b] = (float)s_sum[2];
  }
}

// ---------------------------------------------------------------------
// Transpose + convert weights:  Wt[n][k] = (f16)W[k][n], zero-padded K
// ---------------------------------------------------------------------
__global__ __launch_bounds__(256)
void pack_w_kernel(const float* __restrict__ W, _Float16* __restrict__ Wt,
                   int K, int Kpad) {
  long t = (long)blockIdx.x * 256 + threadIdx.x;
  if (t >= (long)DD * Kpad) return;
  int n = (int)(t / Kpad);
  int k = (int)(t % Kpad);
  Wt[t] = (k < K) ? (_Float16)W[(long)k * DD + n] : (_Float16)0.f;
}

// ---------------------------------------------------------------------
// Gathers into X + initialize LightGCN columns (layer-0 term, x0.25)
// ---------------------------------------------------------------------
__global__ void gather_init_kernel(const int* __restrict__ aux,
                                   const int* __restrict__ uid,
                                   const int* __restrict__ iid,
                                   const float* __restrict__ rate_tab,
                                   const float* __restrict__ gender_tab,
                                   const float* __restrict__ age_tab,
                                   const float* __restrict__ occ_tab,
                                   const float* __restrict__ area_tab,
                                   const float* __restrict__ user_rel,
                                   const float* __restrict__ item_rel,
                                   float* __restrict__ X) {
  int b = blockIdx.x;
  int t = threadIdx.x;
  const int* arow = aux + (long)b * AUX_STRIDE;
  float* xrow = X + (long)b * X_COLS;
  xrow[0   + t] = rate_tab  [(long)arow[0]     * DD + t];
  xrow[256 + t] = gender_tab[(long)arow[10242] * DD + t];
  xrow[320 + t] = age_tab   [(long)arow[10243] * DD + t];
  xrow[384 + t] = occ_tab   [(long)arow[10244] * DD + t];
  xrow[448 + t] = area_tab  [(long)arow[10245] * DD + t];
  xrow[512 + t] = 0.25f * user_rel[(long)uid[b] * DD + t];
  xrow[576 + t] = 0.25f * item_rel[(long)iid[b] * DD + t];
}

// ---------------------------------------------------------------------
// SpMM layer: emb_out[row] += val * emb_in[col]; one wave per edge,
// float2 per lane, global_atomic_add_f32 scatter (tables L2-resident).
// ---------------------------------------------------------------------
__global__ void spmm_kernel(const int* __restrict__ row,
                            const int* __restrict__ col,
                            const float* __restrict__ val,
                            const float* __restrict__ emb_in,
                            float* __restrict__ emb_out,
                            int n_edges) {
  long t = (long)blockIdx.x * blockDim.x + threadIdx.x;
  long e = t >> 5;
  int lane = (int)(t & 31);
  if (e >= n_edges) return;
  int r = row[e], c = col[e];
  float v = val[e];
  const float2* src = (const float2*)(emb_in + (long)c * DD);
  float2 d = src[lane];
  float* dst = emb_out + (long)r * DD + lane * 2;
  atomicAdd(dst,     v * d.x);
  atomicAdd(dst + 1, v * d.y);
}

__global__ void accum_gcn_kernel(const int* __restrict__ uid,
                                 const int* __restrict__ iid,
                                 const float* __restrict__ emb,
                                 float* __restrict__ X) {
  int b = blockIdx.x;
  int t = threadIdx.x;
  X[(long)b * X_COLS + 512 + t] += 0.25f * emb[(long)uid[b] * DD + t];
  X[(long)b * X_COLS + 576 + t] += 0.25f * emb[((long)iid[b] + N_USER) * DD + t];
}

// ---------------------------------------------------------------------
// Multi-hot matmul: Y = (A @ Wt^T) / rsum.
// One wave computes a full 16x64 row strip: A fragment loaded once per
// K-chunk, reused by 4 WMMAs (4 accumulators). Inner loop:
//   2x A b128 + 8x B b128 loads -> 4x v_wmma_f32_16x16x32_f16.
// Unconditional speculative prefetch on the dominant A stream.
// ---------------------------------------------------------------------
__global__ __launch_bounds__(32)
void multihot_wmma_kernel(const _Float16* __restrict__ A,
                          const _Float16* __restrict__ Wt,
                          const float* __restrict__ rsum,
                          int Kpad, float* __restrict__ X, int x_col) {
  int rowTile = blockIdx.x;      // 0..63
  int lane = threadIdx.x;
  int g = lane >> 4;
  int m = lane & 15;
  int row0 = rowTile * 16;

  const _Float16* arow = A + (long)(row0 + m) * Kpad + 16 * g;
  const _Float16* w0 = Wt + (long)(0 * 16 + m) * Kpad + 16 * g;
  const _Float16* w1 = Wt + (long)(1 * 16 + m) * Kpad + 16 * g;
  const _Float16* w2 = Wt + (long)(2 * 16 + m) * Kpad + 16 * g;
  const _Float16* w3 = Wt + (long)(3 * 16 + m) * Kpad + 16 * g;

  v8f c0 = {}, c1 = {}, c2 = {}, c3 = {};
  for (int k0 = 0; k0 < Kpad; k0 += 32) {
    v16h a  = cat8(*(const v8h*)(arow + k0), *(const v8h*)(arow + k0 + 8));
    v16h b0 = cat8(*(const v8h*)(w0 + k0),   *(const v8h*)(w0 + k0 + 8));
    v16h b1 = cat8(*(const v8h*)(w1 + k0),   *(const v8h*)(w1 + k0 + 8));
    v16h b2 = cat8(*(const v8h*)(w2 + k0),   *(const v8h*)(w2 + k0 + 8));
    v16h b3 = cat8(*(const v8h*)(w3 + k0),   *(const v8h*)(w3 + k0 + 8));
    // speculative prefetch: OOB addresses are silently dropped (ISA 10.5)
    __builtin_prefetch(arow + k0 + 256, 0, 1);
    c0 = __builtin_amdgcn_wmma_f32_16x16x32_f16(false, a, false, b0, (short)0, c0, false, false);
    c1 = __builtin_amdgcn_wmma_f32_16x16x32_f16(false, a, false, b1, (short)0, c1, false, false);
    c2 = __builtin_amdgcn_wmma_f32_16x16x32_f16(false, a, false, b2, (short)0, c2, false, false);
    c3 = __builtin_amdgcn_wmma_f32_16x16x32_f16(false, a, false, b3, (short)0, c3, false, false);
  }
#pragma unroll
  for (int r = 0; r < 8; ++r) {
    int M = r + g * 8;
    float inv = 1.0f / rsum[row0 + M];
    float* xr = X + (long)(row0 + M) * X_COLS + x_col + m;
    xr[0]  = c0[r] * inv;
    xr[16] = c1[r] * inv;
    xr[32] = c2[r] * inv;
    xr[48] = c3[r] * inv;
  }
}

// ---------------------------------------------------------------------
// Fused MLP: X(1024x640) @640x64 relu @64x64 relu @64x1 + b
// 4 waves per 16-row tile; B operands pre-transposed f16.
// ---------------------------------------------------------------------
__global__ __launch_bounds__(128)
void mlp_kernel(const float* __restrict__ X,
                const _Float16* __restrict__ fc1t, const float* __restrict__ fc1b,
                const _Float16* __restrict__ fc2t, const float* __restrict__ fc2b,
                const float* __restrict__ outW, const float* __restrict__ outb,
                float* __restrict__ out) {
  int rowTile = blockIdx.x;            // 0..63
  int wave = threadIdx.x >> 5;         // 0..3 -> N tile
  int lane = threadIdx.x & 31;
  int g = lane >> 4;
  int m = lane & 15;
  int row0 = rowTile * 16;

  __shared__ _Float16 s_h1[16][64];
  __shared__ float    s_h2[16][64];

  const float*    xrow = X + (long)(row0 + m) * X_COLS;
  const _Float16* w1   = fc1t + (long)(wave * 16 + m) * X_COLS;
  const _Float16* w2   = fc2t + (long)(wave * 16 + m) * DD;

  // ---- layer 1: K = 640 ----
  v8f c = {};
  for (int k0 = 0; k0 < X_COLS; k0 += 32) {
    v8f xa0 = *(const v8f*)(xrow + k0 + 16 * g);
    v8f xa1 = *(const v8f*)(xrow + k0 + 16 * g + 8);
    v8h blo = *(const v8h*)(w1 + k0 + 16 * g);
    v8h bhi = *(const v8h*)(w1 + k0 + 16 * g + 8);
    v16h a;
#pragma unroll
    for (int e = 0; e < 8; ++e) {
      a[e]     = (_Float16)xa0[e];
      a[8 + e] = (_Float16)xa1[e];
    }
    c = __builtin_amdgcn_wmma_f32_16x16x32_f16(false, a, false, cat8(blo, bhi),
                                               (short)0, c, false, false);
  }
#pragma unroll
  for (int r = 0; r < 8; ++r) {
    int M = r + g * 8;
    float y = c[r] + fc1b[wave * 16 + m];
    s_h1[M][wave * 16 + m] = (_Float16)(y > 0.f ? y : 0.f);
  }
  __syncthreads();

  // ---- layer 2: K = 64 (A from LDS) ----
  v8f c2 = {};
#pragma unroll
  for (int k0 = 0; k0 < DD; k0 += 32) {
    v8h alo = *(const v8h*)(&s_h1[m][k0 + 16 * g]);
    v8h ahi = *(const v8h*)(&s_h1[m][k0 + 16 * g + 8]);
    v8h blo = *(const v8h*)(w2 + k0 + 16 * g);
    v8h bhi = *(const v8h*)(w2 + k0 + 16 * g + 8);
    c2 = __builtin_amdgcn_wmma_f32_16x16x32_f16(false, cat8(alo, ahi), false,
                                                cat8(blo, bhi), (short)0, c2,
                                                false, false);
  }
#pragma unroll
  for (int r = 0; r < 8; ++r) {
    int M = r + g * 8;
    float y = c2[r] + fc2b[wave * 16 + m];
    s_h2[M][wave * 16 + m] = (y > 0.f ? y : 0.f);
  }
  __syncthreads();

  // ---- output layer: per-row dot(64) ----
  int tr = threadIdx.x >> 3;   // row 0..15, 8 threads per row
  int tc = threadIdx.x & 7;
  float p = 0.f;
#pragma unroll
  for (int q = 0; q < 8; ++q) p += s_h2[tr][tc * 8 + q] * outW[tc * 8 + q];
  p += __shfl_down(p, 4, 8);
  p += __shfl_down(p, 2, 8);
  p += __shfl_down(p, 1, 8);
  if (tc == 0) out[row0 + tr] = p + outb[0];
}

// ---------------------------------------------------------------------
static inline void* carve(char** cur, size_t bytes) {
  void* p = (void*)*cur;
  size_t aligned = (bytes + 255) & ~(size_t)255;
  *cur += aligned;
  return p;
}

extern "C" void kernel_launch(void* const* d_in, const int* in_sizes, int n_in,
                              void* d_out, int out_size, void* d_ws, size_t ws_size,
                              hipStream_t stream) {
  const int*   aux       = (const int*)  d_in[0];
  const int*   user_ids  = (const int*)  d_in[1];
  const int*   item_ids  = (const int*)  d_in[2];
  const int*   graph_row = (const int*)  d_in[3];
  const int*   graph_col = (const int*)  d_in[4];
  const float* graph_val = (const float*)d_in[5];
  const float* rate_tab  = (const float*)d_in[6];
  const float* genre_W   = (const float*)d_in[7];
  const float* dir_W     = (const float*)d_in[8];
  const float* act_W     = (const float*)d_in[9];
  const float* gender_tab= (const float*)d_in[10];
  const float* age_tab   = (const float*)d_in[11];
  const float* occ_tab   = (const float*)d_in[12];
  const float* area_tab  = (const float*)d_in[13];
  const float* user_rel  = (const float*)d_in[14];
  const float* item_rel  = (const float*)d_in[15];
  const float* fc1W      = (const float*)d_in[16];
  const float* fc1b      = (const float*)d_in[17];
  const float* fc2W      = (const float*)d_in[18];
  const float* fc2b      = (const float*)d_in[19];
  const float* outW      = (const float*)d_in[20];
  const float* outb      = (const float*)d_in[21];
  float* out = (float*)d_out;
  int n_edges = in_sizes[3];                 // 2E

  // ---- workspace layout ----
  size_t embElems = (size_t)N_NODES * DD;
  char* cur = (char*)d_ws;
  float*    embA = (float*)   carve(&cur, embElems * 4);
  float*    embB = (float*)   carve(&cur, embElems * 4);
  float*    X    = (float*)   carve(&cur, (size_t)B_ROWS * X_COLS * 4);
  _Float16* Ag   = (_Float16*)carve(&cur, (size_t)B_ROWS * KP_GENRE * 2);
  _Float16* Ad   = (_Float16*)carve(&cur, (size_t)B_ROWS * KP_DIR * 2);
  _Float16* Aa   = (_Float16*)carve(&cur, (size_t)B_ROWS * KP_ACT * 2);
  _Float16* Wtg  = (_Float16*)carve(&cur, (size_t)DD * KP_GENRE * 2);
  _Float16* Wtd  = (_Float16*)carve(&cur, (size_t)DD * KP_DIR * 2);
  _Float16* Wta  = (_Float16*)carve(&cur, (size_t)DD * KP_ACT * 2);
  _Float16* fc1t = (_Float16*)carve(&cur, (size_t)DD * X_COLS * 2);
  _Float16* fc2t = (_Float16*)carve(&cur, (size_t)DD * DD * 2);
  float*    rsg  = (float*)   carve(&cur, B_ROWS * 4);
  float*    rsd  = (float*)   carve(&cur, B_ROWS * 4);
  float*    rsa  = (float*)   carve(&cur, B_ROWS * 4);

  // ---- packing passes ----
  pack_aux_kernel<<<B_ROWS, 256, 0, stream>>>(aux, Ag, Ad, Aa, rsg, rsd, rsa);
  auto wblocks = [](long n) { return (int)((n + 255) / 256); };
  pack_w_kernel<<<wblocks((long)DD * KP_GENRE), 256, 0, stream>>>(genre_W, Wtg, K_GENRE, KP_GENRE);
  pack_w_kernel<<<wblocks((long)DD * KP_DIR),   256, 0, stream>>>(dir_W,   Wtd, K_DIR,   KP_DIR);
  pack_w_kernel<<<wblocks((long)DD * KP_ACT),   256, 0, stream>>>(act_W,   Wta, K_ACT,   KP_ACT);
  pack_w_kernel<<<wblocks((long)DD * X_COLS),   256, 0, stream>>>(fc1W, fc1t, X_COLS, X_COLS);
  pack_w_kernel<<<wblocks((long)DD * DD),       256, 0, stream>>>(fc2W, fc2t, DD, DD);

  // ---- gathers + LightGCN column init ----
  gather_init_kernel<<<B_ROWS, 64, 0, stream>>>(aux, user_ids, item_ids,
      rate_tab, gender_tab, age_tab, occ_tab, area_tab, user_rel, item_rel, X);

  // ---- LightGCN: emb0 then 3 propagation layers ----
  hipMemcpyAsync(embA, user_rel, (size_t)N_USER * DD * sizeof(float),
                 hipMemcpyDeviceToDevice, stream);
  hipMemcpyAsync(embA + (size_t)N_USER * DD, item_rel,
                 (size_t)M_ITEM * DD * sizeof(float),
                 hipMemcpyDeviceToDevice, stream);
  float* curp = embA;
  float* nxtp = embB;
  long spmm_threads = (long)n_edges * 32;
  int  spmm_blocks  = (int)((spmm_threads + 255) / 256);
  for (int layer = 0; layer < 3; ++layer) {
    hipMemsetAsync(nxtp, 0, embElems * sizeof(float), stream);
    spmm_kernel<<<spmm_blocks, 256, 0, stream>>>(graph_row, graph_col, graph_val,
                                                 curp, nxtp, n_edges);
    accum_gcn_kernel<<<B_ROWS, 64, 0, stream>>>(user_ids, item_ids, nxtp, X);
    float* tmp = curp; curp = nxtp; nxtp = tmp;
  }

  // ---- multi-hot WMMA matmuls -> X cols 64..256 (16x64 strip per wave) ----
  multihot_wmma_kernel<<<B_ROWS / 16, 32, 0, stream>>>(Ag, Wtg, rsg, KP_GENRE, X, 64);
  multihot_wmma_kernel<<<B_ROWS / 16, 32, 0, stream>>>(Ad, Wtd, rsd, KP_DIR,   X, 128);
  multihot_wmma_kernel<<<B_ROWS / 16, 32, 0, stream>>>(Aa, Wta, rsa, KP_ACT,   X, 192);

  // ---- fused MLP ----
  mlp_kernel<<<B_ROWS / 16, 128, 0, stream>>>(X, fc1t, fc1b, fc2t, fc2b,
                                              outW, outb, out);
}
